// RNNLayer_50732153700771
// MI455X (gfx1250) — matrix operand order repaired and checked by reading, not compiled, not run on hardware
//
#include <hip/hip_runtime.h>
#include <hip/hip_bf16.h>

// ---------------- problem constants ----------------
#define TT   1024
#define BB   32
#define II   512
#define HH   512
#define G4H  2048            // 4*H
#define NWG  64              // persistent workgroups in recurrence
#define JPW  8               // hidden columns owned per WG (HH / NWG)

// ---------------- d_out layout (floats) ----------------
// output [T,B,H] | h_T [1,B,H] | c_T [1,B,H] | all_hidden [T,B,H]
static constexpr size_t OUT_Y    = 0;
static constexpr size_t OUT_HT   = (size_t)TT * BB * HH;            // 16777216
static constexpr size_t OUT_CT   = OUT_HT + (size_t)BB * HH;        // 16793600
static constexpr size_t OUT_ALLH = OUT_CT + (size_t)BB * HH;        // 16809984

// ---------------- workspace layout (bytes) ----------------
static constexpr size_t WS_GX   = 0;                                  // 32768*2048*4 = 256 MB
static constexpr size_t WS_XBF  = WS_GX   + (size_t)TT*BB*G4H*4;      // x bf16, 32 MB
static constexpr size_t WS_WXBF = WS_XBF  + (size_t)TT*BB*II*2;       // Wx bf16, 2 MB
static constexpr size_t WS_WHBF = WS_WXBF + (size_t)G4H*II*2;         // Wh bf16, 2 MB
static constexpr size_t WS_HBF  = WS_WHBF + (size_t)G4H*HH*2;         // h double buffer bf16
static constexpr size_t WS_CTR  = WS_HBF  + (size_t)2*BB*HH*2;        // barrier counter

// ---------------- types ----------------
typedef __bf16 bf16_t;
typedef __attribute__((ext_vector_type(16))) __bf16 v16bf;
typedef __attribute__((ext_vector_type(8)))  __bf16 v8bf;
typedef __attribute__((ext_vector_type(8)))  float  v8f;

__device__ __forceinline__ bf16_t f2bf(float f) {
    unsigned u = __builtin_bit_cast(unsigned, f);
    u = u + 0x7FFFu + ((u >> 16) & 1u);          // round-to-nearest-even
    unsigned short s = (unsigned short)(u >> 16);
    return __builtin_bit_cast(bf16_t, s);
}

__device__ __forceinline__ v16bf cat8(v8bf lo, v8bf hi) {
    return __builtin_shufflevector(lo, hi, 0,1,2,3,4,5,6,7,8,9,10,11,12,13,14,15);
}

__device__ __forceinline__ v8f wmma_bf16(v16bf a, v16bf b, v8f c) {
    // (neg_a, A, neg_b, B, c_mod, C, reuse_a, reuse_b)
    return __builtin_amdgcn_wmma_f32_16x16x32_bf16(false, a, false, b, (short)0, c,
                                                   false, false);
}

__device__ __forceinline__ float fsigmoid(float x) {
    return 1.0f / (1.0f + __expf(-x));
}

// ---------------- helper kernels ----------------
__global__ void f32_to_bf16(const float* __restrict__ src, bf16_t* __restrict__ dst, int n) {
    for (int i = blockIdx.x * blockDim.x + threadIdx.x; i < n; i += gridDim.x * blockDim.x)
        dst[i] = f2bf(src[i]);
}

__global__ void init_state(const float* __restrict__ h0, bf16_t* __restrict__ hbf,
                           unsigned* __restrict__ ctr) {
    int i = blockIdx.x * blockDim.x + threadIdx.x;
    if (i < BB * HH) hbf[i] = f2bf(h0[i]);       // buffer 0 := h0
    if (i == 0)      *ctr = 0u;                  // reset grid barrier (per launch)
}

// ---------------- phase 1: Gx = X @ Wx^T + b  (bf16 WMMA, fp32 acc) ----------------
// M = T*B = 32768, N = 4H = 2048, K = I = 512.
// One wave computes a 32x64 strip (2 M-tiles x 4 N-tiles): each B fragment feeds
// 2 wmmas, each A fragment feeds 4. All 12 b128 loads of a k-slice are issued
// before the 8 wmmas so the scheduler gets a deep clause to hide load latency.
__global__ __launch_bounds__(256) void gemm_gates_x(
    const bf16_t* __restrict__ Xbf,   // [32768][512] row-major
    const bf16_t* __restrict__ Wxbf,  // [2048][512]  row-major (== B^T, contiguous in K)
    const float*  __restrict__ bias,  // [2048]
    float*        __restrict__ G)     // [32768][2048]
{
    const int lane = threadIdx.x & 31;
    const int gw   = blockIdx.x * 8 + (threadIdx.x >> 5);
    const int mb   = gw & 1023;          // 1024 M-blocks of 32 rows
    const int ng   = gw >> 10;           // 32 groups of 64 N-columns
    const int m0   = mb << 5;
    const int n0   = ng << 6;

    const bf16_t* Arow0 = Xbf + (size_t)(m0 + (lane & 15)) * II + ((lane >> 4) << 3);
    const bf16_t* Arow1 = Arow0 + (size_t)16 * II;
    const bf16_t* Brow[4];
#pragma unroll
    for (int j = 0; j < 4; ++j)
        Brow[j] = Wxbf + (size_t)(n0 + j * 16 + (lane & 15)) * II + ((lane >> 4) << 3);

    v8f acc[2][4] = {};
    for (int kk = 0; kk < II; kk += 32) {
        // ---- issue every load of this k-slice first ----
        v8bf a0lo = *(const v8bf*)(Arow0 + kk);
        v8bf a0hi = *(const v8bf*)(Arow0 + kk + 16);
        v8bf a1lo = *(const v8bf*)(Arow1 + kk);
        v8bf a1hi = *(const v8bf*)(Arow1 + kk + 16);
        v8bf blo[4], bhi[4];
#pragma unroll
        for (int j = 0; j < 4; ++j) {
            blo[j] = *(const v8bf*)(Brow[j] + kk);
            bhi[j] = *(const v8bf*)(Brow[j] + kk + 16);
        }
        v16bf a0 = cat8(a0lo, a0hi);
        v16bf a1 = cat8(a1lo, a1hi);
        // ---- 8 wmmas, B fragments reused across the two M-tiles ----
#pragma unroll
        for (int j = 0; j < 4; ++j) {
            v16bf bj = cat8(blo[j], bhi[j]);
            acc[0][j] = wmma_bf16(a0, bj, acc[0][j]);
            acc[1][j] = wmma_bf16(a1, bj, acc[1][j]);
        }
    }

    const int ncol = n0 + (lane & 15);
#pragma unroll
    for (int i = 0; i < 2; ++i) {
        const int mrow = m0 + i * 16 + ((lane >> 4) << 3);
#pragma unroll
        for (int j = 0; j < 4; ++j) {
            const int c = ncol + j * 16;
            const float bc = bias[c];
#pragma unroll
            for (int r = 0; r < 8; ++r)
                G[(size_t)(mrow + r) * G4H + c] = acc[i][j][r] + bc;
        }
    }
}

// ---------------- phase 2: persistent LSTM recurrence ----------------
// 64 WGs x 128 threads (4 waves). WG `wg` owns hidden columns [wg*8, wg*8+8):
// it computes the 32 gate columns {g*H + wg*8 + jj : g in 0..3, jj in 0..7},
// holding its Wh slice (32 rows x 512, bf16) in LDS for all 1024 steps.
// h is exchanged through L2 (bf16, double-buffered) with a monotonic-counter
// grid barrier each step; c lives in LDS for the whole kernel. The next step's
// Gx slice (cold HBM stream) is prefetched behind the current wmma loop.
__global__ __launch_bounds__(128) void lstm_rec(
    const float*  __restrict__ c0,    // [B*H]
    const float*  __restrict__ Gx,    // [T*B][2048]
    const bf16_t* __restrict__ Whbf,  // [2048][512]
    bf16_t*       __restrict__ hbf,   // [2][B*H]
    float*        __restrict__ out,
    unsigned*     __restrict__ ctr)
{
    __shared__ bf16_t Whs[32][II];        // 32 KB: local gate-col l -> Wh row
    __shared__ float  gates_s[BB][32];    // 4 KB
    __shared__ float  c_s[BB][JPW];       // 1 KB

    const int wg   = blockIdx.x;
    const int tid  = threadIdx.x;
    const int lane = tid & 31;
    const int wave = tid >> 5;
    const int mt   = wave & 1;            // rows  [mt*16, mt*16+16) of B=32
    const int nt   = wave >> 1;           // local cols [nt*16, nt*16+16) of 32

    // Load Wh slice into LDS: local row l -> Wh[(l>>3)*H + wg*8 + (l&7)].
    for (int c = tid; c < 32 * (II / 8); c += 128) {
        int l    = c >> 6;                // 64 8-element chunks per row
        int kc   = (c & 63) << 3;
        int grow = ((l >> 3) * HH) + wg * JPW + (l & 7);
        *(v8bf*)&Whs[l][kc] = *(const v8bf*)(Whbf + (size_t)grow * II + kc);
    }
    for (int i = tid; i < BB * JPW; i += 128) {
        int b = i >> 3, jj = i & 7;
        c_s[b][jj] = c0[b * HH + wg * JPW + jj];
    }
    __syncthreads();

    for (int t = 0; t < TT; ++t) {
        // Prefetch next step's gate slice (128 cachelines across 128 threads)
        // so the HBM latency overlaps this step's wmma loop.
        if (t + 1 < TT) {
            const float* pf = Gx + (size_t)((t + 1) * BB + (tid >> 2)) * G4H
                                 + (tid & 3) * HH + wg * JPW;
            __builtin_prefetch(pf, 0, 1);
        }

        const bf16_t* hcur  = hbf + (size_t)(t & 1) * (BB * HH);
        bf16_t*       hnext = hbf + (size_t)((t + 1) & 1) * (BB * HH);

        // gates_rec = h @ Wh_slice^T : one 16x16 tile per wave, K = 512.
        const bf16_t* Arow = hcur + (size_t)(mt * 16 + (lane & 15)) * HH
                                   + ((lane >> 4) << 3);
        v8f acc = {};
#pragma unroll 4
        for (int kk = 0; kk < HH; kk += 32) {
            v8bf alo = *(const v8bf*)(Arow + kk);
            v8bf ahi = *(const v8bf*)(Arow + kk + 16);
            int  lrow = nt * 16 + (lane & 15);
            int  kb   = kk + ((lane >> 4) << 3);
            v8bf blo = *(const v8bf*)&Whs[lrow][kb];
            v8bf bhi = *(const v8bf*)&Whs[lrow][kb + 16];
            acc = wmma_bf16(cat8(alo, ahi), cat8(blo, bhi), acc);
        }

        // Add precomputed x-gates and stage into LDS.
        {
            const int bb0  = mt * 16 + ((lane >> 4) << 3);
            const int l    = nt * 16 + (lane & 15);
            const int gcol = ((l >> 3) * HH) + wg * JPW + (l & 7);
            const float* gx = Gx + (size_t)(t * BB) * G4H + gcol;
#pragma unroll
            for (int r = 0; r < 8; ++r)
                gates_s[bb0 + r][l] = acc[r] + gx[(size_t)(bb0 + r) * G4H];
        }
        __syncthreads();

        // Gate nonlinearities + state update (torch order i, f, g, o).
        for (int i = tid; i < BB * JPW; i += 128) {
            int b = i >> 3, jj = i & 7;
            float si = fsigmoid(gates_s[b][jj]);
            float sf = fsigmoid(gates_s[b][8  + jj]);
            float gg = tanhf   (gates_s[b][16 + jj]);
            float so = fsigmoid(gates_s[b][24 + jj]);
            float cn = sf * c_s[b][jj] + si * gg;
            c_s[b][jj] = cn;
            float h = so * tanhf(cn);
            int col  = wg * JPW + jj;
            size_t o = (size_t)t * (BB * HH) + (size_t)b * HH + col;
            // write-once streams: keep them out of the caches (TH=NT)
            __builtin_nontemporal_store(h, &out[OUT_Y    + o]);
            __builtin_nontemporal_store(h, &out[OUT_ALLH + o]);
            hnext[b * HH + col] = f2bf(h);
            if (t == TT - 1) {
                out[OUT_HT + b * HH + col] = h;
                out[OUT_CT + b * HH + col] = cn;
            }
        }

        // Grid barrier: release own h writes, arrive, spin, acquire.
        __threadfence();
        __syncthreads();
        if (tid == 0) {
            __hip_atomic_fetch_add(ctr, 1u, __ATOMIC_RELEASE, __HIP_MEMORY_SCOPE_AGENT);
            const unsigned target = (unsigned)(t + 1) * NWG;
            while (__hip_atomic_load(ctr, __ATOMIC_ACQUIRE, __HIP_MEMORY_SCOPE_AGENT) < target)
                __builtin_amdgcn_s_sleep(2);
        }
        __syncthreads();
        __builtin_amdgcn_fence(__ATOMIC_ACQUIRE, "agent");
    }
}

// ---------------- launcher ----------------
extern "C" void kernel_launch(void* const* d_in, const int* in_sizes, int n_in,
                              void* d_out, int out_size, void* d_ws, size_t ws_size,
                              hipStream_t stream) {
    (void)in_sizes; (void)n_in; (void)out_size; (void)ws_size;

    const float* x    = (const float*)d_in[0];   // [T,B,I]
    const float* h0   = (const float*)d_in[1];   // [1,B,H]
    const float* c0   = (const float*)d_in[2];   // [1,B,H]
    const float* Wx   = (const float*)d_in[3];   // [4H,I]
    const float* Wh   = (const float*)d_in[4];   // [4H,H]
    const float* bias = (const float*)d_in[5];   // [4H]

    char* ws = (char*)d_ws;
    float*    G     = (float*)   (ws + WS_GX);
    bf16_t*   xbf   = (bf16_t*)  (ws + WS_XBF);
    bf16_t*   wxbf  = (bf16_t*)  (ws + WS_WXBF);
    bf16_t*   whbf  = (bf16_t*)  (ws + WS_WHBF);
    bf16_t*   hbf   = (bf16_t*)  (ws + WS_HBF);
    unsigned* ctr   = (unsigned*)(ws + WS_CTR);

    float* out = (float*)d_out;

    // bf16 copies of x, Wx, Wh
    f32_to_bf16<<<4096, 256, 0, stream>>>(x,  xbf,  TT * BB * II);
    f32_to_bf16<<<256,  256, 0, stream>>>(Wx, wxbf, G4H * II);
    f32_to_bf16<<<256,  256, 0, stream>>>(Wh, whbf, G4H * HH);

    // h double-buffer[0] := h0, barrier counter := 0 (every launch -> graph-safe)
    init_state<<<(BB * HH + 255) / 256, 256, 0, stream>>>(h0, hbf, ctr);

    // Phase 1: Gx = X @ Wx^T + b   (32768 waves, 8 per block, 32x64 per wave)
    gemm_gates_x<<<4096, 256, 0, stream>>>(xbf, wxbf, bias, G);

    // Phase 2: persistent recurrence
    lstm_rec<<<NWG, 128, 0, stream>>>(c0, G, whbf, hbf, out, ctr);
}
